// MHA_5660766896669
// MI455X (gfx1250) — matrix operand order repaired and checked by reading, not compiled, not run on hardware
//
#include <hip/hip_runtime.h>
#include <hip/hip_bf16.h>

// ---------------------------------------------------------------------------
// Causal linear-attention block for MI455X (gfx1250).
// bf16 WMMA matrix core + async global->LDS staging (CDNA5 ASYNCcnt path).
// B=2, S=2048, D_MODEL=512, H=8, D_HEAD=64.
// ---------------------------------------------------------------------------

typedef __attribute__((ext_vector_type(16))) __bf16 v16bf;
typedef __attribute__((ext_vector_type(8)))  float  v8f;

#define BB     2
#define SS     2048
#define DM     512
#define HH     8
#define DH     64
#define MROWS  (BB * SS)      // 4096
#define NQKV   (2 * DM + DM)  // 1536
#define LDT    72             // padded LDS tile stride (bf16 elems); keeps 16B align

// ---- CDNA5 async global->LDS copy (16 bytes per lane per issue) ------------
__device__ inline void async_copy_b128(unsigned lds_off, const void* gptr) {
    asm volatile("global_load_async_to_lds_b128 %0, %1, off"
                 :: "v"(lds_off), "v"((unsigned long long)(size_t)gptr)
                 : "memory");
}
__device__ inline void async_wait0() {
    asm volatile("s_wait_asynccnt 0x0" ::: "memory");
}
// LDS byte offset of a __shared__ object (generic addr low 32 bits = LDS addr).
#define LDS_OFF(p) ((unsigned)(size_t)(p))

// ---- WMMA fragment gathers (layouts per CDNA5 ISA 7.12.2) ------------------

// A operand 16x32 bf16 from a row-major bf16 LDS tile.
__device__ inline v16bf gatherA_bf(const __bf16* t, int ld, int rowBase, int k0, int lane) {
    int r = rowBase + (lane & 15);
    int half = lane >> 4;
    const __bf16* p = t + r * ld + k0;
    v16bf a;
#pragma unroll
    for (int i = 0; i < 8; ++i) a[i] = p[8 * half + i];
#pragma unroll
    for (int i = 0; i < 8; ++i) a[8 + i] = p[16 + 8 * half + i];
    return a;
}

// Same gather but source tile is f32 (convert to bf16 per element).
__device__ inline v16bf gatherA_f32(const float* t, int ld, int rowBase, int k0, int lane) {
    int r = rowBase + (lane & 15);
    int half = lane >> 4;
    const float* p = t + r * ld + k0;
    v16bf a;
#pragma unroll
    for (int i = 0; i < 8; ++i) a[i] = (__bf16)p[8 * half + i];
#pragma unroll
    for (int i = 0; i < 8; ++i) a[8 + i] = (__bf16)p[16 + 8 * half + i];
    return a;
}

// B operand 32x16 bf16 from a B-TRANSPOSED ([n][k] row-major) bf16 LDS tile.
__device__ inline v16bf gatherB_bf(const __bf16* t, int ld, int nBase, int k0, int lane) {
    int n = nBase + (lane & 15);
    int half = lane >> 4;
    const __bf16* p = t + n * ld + k0;
    v16bf b;
#pragma unroll
    for (int i = 0; i < 8; ++i) b[i] = p[8 * half + i];
#pragma unroll
    for (int i = 0; i < 8; ++i) b[8 + i] = p[16 + 8 * half + i];
    return b;
}

#define WMMA_BF16(a, b, c) \
    __builtin_amdgcn_wmma_f32_16x16x32_bf16(false, (a), false, (b), (short)0, (c), false, false)

// ---------------------------------------------------------------------------
// Kernel 1: fused QKV projection.
//   [4096 x 512] (x, f32->bf16) @ [512 x 1536] (concat(Wqk;Wv)^T) + bias
// Epilogue: elu+1 on q,k; scatter Q,K as [b,h,s,d] bf16 and V^T as [b,h,d,s].
// ---------------------------------------------------------------------------
__global__ void __launch_bounds__(128)
qkv_proj_kernel(const float* __restrict__ x,
                const float* __restrict__ Wqk_w, const float* __restrict__ Wqk_b,
                const float* __restrict__ Wv_w,  const float* __restrict__ Wv_b,
                __bf16* __restrict__ Qb, __bf16* __restrict__ Kb,
                __bf16* __restrict__ vT) {
    __shared__ __bf16 As[64][LDT];
    __shared__ __bf16 Bs[64][LDT];

    const int tid  = threadIdx.x;
    const int w    = tid >> 5;
    const int lane = tid & 31;
    const int n0   = blockIdx.x * 64;
    const int m0   = blockIdx.y * 64;

    // Staging role: each thread owns a contiguous 32-float run (vectorizes to b128 loads).
    const int srow  = tid >> 1;
    const int scol  = (tid & 1) * 32;

    v8f acc[4] = {};

    for (int k0 = 0; k0 < DM; k0 += 64) {
        {
            const float* asrc = x + (size_t)(m0 + srow) * DM + k0 + scol;
            __bf16* adst = &As[srow][scol];
#pragma unroll
            for (int i = 0; i < 32; ++i) adst[i] = (__bf16)asrc[i];

            int gn = n0 + srow;
            const float* bsrc = (gn < 2 * DM)
                ? (Wqk_w + (size_t)gn * DM + k0 + scol)
                : (Wv_w + (size_t)(gn - 2 * DM) * DM + k0 + scol);
            __bf16* bdst = &Bs[srow][scol];
#pragma unroll
            for (int i = 0; i < 32; ++i) bdst[i] = (__bf16)bsrc[i];
        }
        __syncthreads();
#pragma unroll
        for (int kk = 0; kk < 64; kk += 32) {
            v16bf a = gatherA_bf(&As[0][0], LDT, w * 16, kk, lane);
#pragma unroll
            for (int nt = 0; nt < 4; ++nt) {
                v16bf b = gatherB_bf(&Bs[0][0], LDT, nt * 16, kk, lane);
                acc[nt] = WMMA_BF16(a, b, acc[nt]);
            }
        }
        __syncthreads();
    }

    const int col  = lane & 15;
    const int half = lane >> 4;
#pragma unroll
    for (int nt = 0; nt < 4; ++nt) {
        int gn = n0 + nt * 16 + col;
        float bias = (gn < 2 * DM) ? Wqk_b[gn] : Wv_b[gn - 2 * DM];
#pragma unroll
        for (int r = 0; r < 8; ++r) {
            int gm = m0 + w * 16 + r + 8 * half;
            int bb = gm >> 11;          // / S
            int s  = gm & (SS - 1);     // % S
            float val = acc[nt][r] + bias;
            if (gn < 2 * DM) {
                float f = (val > 0.0f) ? (val + 1.0f) : __expf(val);  // elu(x)+1
                int e   = gn >> 9;       // 0=q, 1=k
                int rem = gn & (DM - 1);
                int h   = rem >> 6;
                int d   = rem & 63;
                __bf16* dst = (e == 0) ? Qb : Kb;
                dst[(((size_t)(bb * HH + h)) * SS + s) * DH + d] = (__bf16)f;
            } else {
                int vn = gn - 2 * DM;
                int h  = vn >> 6;
                int d  = vn & 63;
                vT[(((size_t)(bb * HH + h)) * DH + d) * SS + s] = (__bf16)val;
            }
        }
    }
}

// ---------------------------------------------------------------------------
// Kernel 2: causal attention with row-sum normalization (flash-style tiles).
// Q/K/V^T tiles staged via CDNA5 async global->LDS DMA.
// grid (S/64, H, B), block 128 (4 waves); wave w owns query rows w*16..+15.
// ---------------------------------------------------------------------------
__global__ void __launch_bounds__(128)
attention_kernel(const __bf16* __restrict__ Q, const __bf16* __restrict__ K,
                 const __bf16* __restrict__ vT, __bf16* __restrict__ ctx) {
    __shared__ __bf16 Qs[64][LDT];
    __shared__ __bf16 Ks[64][LDT];
    __shared__ __bf16 Vts[64][LDT];         // V^T tile: [d][s_local]
    __shared__ float  Pf[4][16][64];        // per-wave unnormalized probs (f32)
    __shared__ float  nuS[64];

    const int tid  = threadIdx.x;
    const int w    = tid >> 5;
    const int lane = tid & 31;
    const int qb   = blockIdx.x;
    const int h    = blockIdx.y;
    const int bz   = blockIdx.z;

    const __bf16* Qg  = Q  + ((size_t)(bz * HH + h)) * SS * DH;
    const __bf16* Kg  = K  + ((size_t)(bz * HH + h)) * SS * DH;
    const __bf16* vTg = vT + ((size_t)(bz * HH + h)) * DH * SS;

    // Async-stage this block's 64 query rows once (8KB = 4 b128 issues/thread).
    {
        unsigned qbase = LDS_OFF(&Qs[0][0]);
        for (int e = tid; e < 512; e += 128) {
            int r = e >> 3, c8 = (e & 7) * 8;
            async_copy_b128(qbase + (unsigned)(r * LDT + c8) * 2,
                            Qg + (size_t)(qb * 64 + r) * DH + c8);
        }
        async_wait0();
    }

    v8f   cacc[4] = {};
    float rsum    = 0.0f;
    const int col  = lane & 15;
    const int half = lane >> 4;

    for (int j = 0; j <= qb; ++j) {
        __syncthreads();   // previous tiles fully consumed / Qs visible
        {
            unsigned kbase = LDS_OFF(&Ks[0][0]);
            unsigned vbase = LDS_OFF(&Vts[0][0]);
            for (int e = tid; e < 512; e += 128) {
                int r = e >> 3, c8 = (e & 7) * 8;
                async_copy_b128(kbase + (unsigned)(r * LDT + c8) * 2,
                                Kg + (size_t)(j * 64 + r) * DH + c8);
                async_copy_b128(vbase + (unsigned)(r * LDT + c8) * 2,
                                vTg + (size_t)r * SS + j * 64 + c8);
            }
            async_wait0();
        }
        if (j < qb)
            __builtin_prefetch(Kg + (size_t)((j + 1) * 64) * DH, 0, 1);
        __syncthreads();

        // S strip (16x64) = Q_strip @ K_tile^T  (K's [s][d] layout == B-transposed)
        v8f sacc[4] = {};
#pragma unroll
        for (int kk = 0; kk < 64; kk += 32) {
            v16bf a = gatherA_bf(&Qs[0][0], LDT, w * 16, kk, lane);
#pragma unroll
            for (int nt = 0; nt < 4; ++nt) {
                v16bf b = gatherB_bf(&Ks[0][0], LDT, nt * 16, kk, lane);
                sacc[nt] = WMMA_BF16(a, b, sacc[nt]);
            }
        }

        // Causal mask + park unnormalized probs in LDS (f32).
#pragma unroll
        for (int nt = 0; nt < 4; ++nt) {
#pragma unroll
            for (int r = 0; r < 8; ++r) {
                int qrow = qb * 64 + w * 16 + r + 8 * half;
                int kcol = j * 64 + nt * 16 + col;
                float v = (kcol <= qrow) ? sacc[nt][r] : 0.0f;
                Pf[w][r + 8 * half][nt * 16 + col] = v;
            }
        }
        __syncthreads();

        // Row-sum accumulation: 2 lanes per row, 32 cols each.
        {
            int   rr = lane & 15;
            int   cb = half * 32;
            float p  = 0.0f;
#pragma unroll
            for (int c = 0; c < 32; ++c) p += Pf[w][rr][cb + c];
            rsum += p;
        }

        // context strip += P_strip (f32 LDS -> bf16 A frag) @ V_tile (B = V^T).
#pragma unroll
        for (int kk = 0; kk < 64; kk += 32) {
            v16bf a = gatherA_f32(&Pf[w][0][0], 64, 0, kk, lane);
#pragma unroll
            for (int nt = 0; nt < 4; ++nt) {
                v16bf b = gatherB_bf(&Vts[0][0], LDT, nt * 16, kk, lane);
                cacc[nt] = WMMA_BF16(a, b, cacc[nt]);
            }
        }
    }

    // Finalize nu per row and normalize.
    __syncthreads();
    float tot = rsum + __shfl_xor(rsum, 16, 32);
    if (lane < 16) nuS[w * 16 + lane] = tot;
    __syncthreads();

#pragma unroll
    for (int nt = 0; nt < 4; ++nt) {
#pragma unroll
        for (int r = 0; r < 8; ++r) {
            int   row = r + 8 * half;
            float nu  = nuS[w * 16 + row];
            float val = cacc[nt][r] / nu;
            size_t gm = (size_t)bz * SS + qb * 64 + w * 16 + row;
            ctx[gm * DM + h * DH + nt * 16 + col] = (__bf16)val;
        }
    }
}

// ---------------------------------------------------------------------------
// Kernel 3: output projection. out = ctx(bf16, async-staged) @ Wo^T + b -> f32.
// ---------------------------------------------------------------------------
__global__ void __launch_bounds__(128)
out_proj_kernel(const __bf16* __restrict__ ctx,
                const float* __restrict__ Wo_w, const float* __restrict__ Wo_b,
                float* __restrict__ out) {
    __shared__ __bf16 As[64][LDT];
    __shared__ __bf16 Bs[64][LDT];

    const int tid  = threadIdx.x;
    const int w    = tid >> 5;
    const int lane = tid & 31;
    const int n0   = blockIdx.x * 64;
    const int m0   = blockIdx.y * 64;

    const int srow = tid >> 1;
    const int scol = (tid & 1) * 32;

    v8f acc[4] = {};

    for (int k0 = 0; k0 < DM; k0 += 64) {
        // A tile: bf16 copy -> async DMA.  B tile: f32->bf16 convert (vector loads).
        {
            unsigned abase = LDS_OFF(&As[0][0]);
            for (int e = tid; e < 512; e += 128) {
                int r = e >> 3, c8 = (e & 7) * 8;
                async_copy_b128(abase + (unsigned)(r * LDT + c8) * 2,
                                ctx + (size_t)(m0 + r) * DM + k0 + c8);
            }
            const float* bsrc = Wo_w + (size_t)(n0 + srow) * DM + k0 + scol;
            __bf16* bdst = &Bs[srow][scol];
#pragma unroll
            for (int i = 0; i < 32; ++i) bdst[i] = (__bf16)bsrc[i];
            async_wait0();
        }
        __syncthreads();
#pragma unroll
        for (int kk = 0; kk < 64; kk += 32) {
            v16bf a = gatherA_bf(&As[0][0], LDT, w * 16, kk, lane);
#pragma unroll
            for (int nt = 0; nt < 4; ++nt) {
                v16bf b = gatherB_bf(&Bs[0][0], LDT, nt * 16, kk, lane);
                acc[nt] = WMMA_BF16(a, b, acc[nt]);
            }
        }
        __syncthreads();
    }

    const int col  = lane & 15;
    const int half = lane >> 4;
#pragma unroll
    for (int nt = 0; nt < 4; ++nt) {
        int gn = n0 + nt * 16 + col;
        float bias = Wo_b[gn];
#pragma unroll
        for (int r = 0; r < 8; ++r) {
            int gm = m0 + w * 16 + r + 8 * half;
            out[(size_t)gm * DM + gn] = acc[nt][r] + bias;
        }
    }
}

// ---------------------------------------------------------------------------
extern "C" void kernel_launch(void* const* d_in, const int* in_sizes, int n_in,
                              void* d_out, int out_size, void* d_ws, size_t ws_size,
                              hipStream_t stream) {
    const float* x     = (const float*)d_in[0];
    const float* Wqk_w = (const float*)d_in[1];
    const float* Wqk_b = (const float*)d_in[2];
    const float* Wv_w  = (const float*)d_in[3];
    const float* Wv_b  = (const float*)d_in[4];
    const float* Wo_w  = (const float*)d_in[5];
    const float* Wo_b  = (const float*)d_in[6];
    float*       out   = (float*)d_out;

    char* ws = (char*)d_ws;
    const size_t SEG = (size_t)BB * HH * SS * DH * sizeof(__bf16); // 4 MiB
    __bf16* Qb  = (__bf16*)(ws);
    __bf16* Kb  = (__bf16*)(ws + SEG);
    __bf16* vT  = (__bf16*)(ws + 2 * SEG);
    __bf16* ctx = (__bf16*)(ws + 3 * SEG);

    qkv_proj_kernel<<<dim3(NQKV / 64, MROWS / 64), 128, 0, stream>>>(
        x, Wqk_w, Wqk_b, Wv_w, Wv_b, Qb, Kb, vT);
    attention_kernel<<<dim3(SS / 64, HH, BB), 128, 0, stream>>>(Qb, Kb, vT, ctx);
    out_proj_kernel<<<dim3(DM / 64, MROWS / 64), 128, 0, stream>>>(
        ctx, Wo_w, Wo_b, out);
}
